// GptOssAttention_76656576299583
// MI455X (gfx1250) — matrix operand order repaired and checked by reading, not compiled, not run on hardware
//
#include <hip/hip_runtime.h>
#include <hip/hip_bf16.h>
#include <stdint.h>

// ---------------------------------------------------------------------------
// Types for CDNA5 WMMA (wave32, 16x16x32 bf16 -> f32)
// ---------------------------------------------------------------------------
typedef __bf16 v16bf __attribute__((ext_vector_type(16)));
typedef float  v8f   __attribute__((ext_vector_type(8)));
typedef unsigned int u32x4 __attribute__((ext_vector_type(4)));

union V16 {
    v16bf v;
    unsigned short u[16];
    u32x4 q[2];
};

__device__ __forceinline__ unsigned short f2bf(float f) {
    unsigned int u = __float_as_uint(f);
    unsigned int r = u + 0x7FFFu + ((u >> 16) & 1u);   // round-to-nearest-even
    return (unsigned short)(r >> 16);
}
__device__ __forceinline__ float bf2f(unsigned short h) {
    return __uint_as_float(((unsigned int)h) << 16);
}

__device__ __forceinline__ v8f wmma_bf16(const V16& a, const V16& b, v8f c) {
    return __builtin_amdgcn_wmma_f32_16x16x32_bf16(
        /*neg_a=*/false, a.v, /*neg_b=*/false, b.v,
        /*c_mod=*/(short)0, c, /*reuse_a=*/false, /*reuse_b=*/false);
}

// ---------------------------------------------------------------------------
// Elementwise helpers: fp32 -> (bf16_hi, bf16_lo) split, optional transpose
// ---------------------------------------------------------------------------
__global__ void split_bf16_kernel(const float* __restrict__ x,
                                  unsigned short* __restrict__ hi,
                                  unsigned short* __restrict__ lo, long n) {
    long i = (long)blockIdx.x * blockDim.x + threadIdx.x;
    if (i >= n) return;
    float f = x[i];
    unsigned short h = f2bf(f);
    hi[i] = h;
    lo[i] = f2bf(f - bf2f(h));
}

// w: [R][C] fp32 row-major  ->  hi/lo: [C][R] bf16 (i.e. W^T)
__global__ void split_bf16_T_kernel(const float* __restrict__ w,
                                    unsigned short* __restrict__ hi,
                                    unsigned short* __restrict__ lo,
                                    int R, int C) {
    long i = (long)blockIdx.x * blockDim.x + threadIdx.x;
    long n = (long)R * C;
    if (i >= n) return;
    int r = (int)(i / C);
    int c = (int)(i - (long)r * C);
    float f = w[i];
    unsigned short h = f2bf(f);
    size_t o = (size_t)c * R + r;
    hi[o] = h;
    lo[o] = f2bf(f - bf2f(h));
}

// RoPE (rotate-half) + bf16 split. X: [S][nh*64] fp32, cos/sin: [S][64]
__global__ void rope_split_kernel(const float* __restrict__ X,
                                  const float* __restrict__ cs,
                                  const float* __restrict__ sn,
                                  unsigned short* __restrict__ hi,
                                  unsigned short* __restrict__ lo,
                                  int S, int nh) {
    long i = (long)blockIdx.x * blockDim.x + threadIdx.x;
    long n = (long)S * nh * 32;
    if (i >= n) return;
    int d = (int)(i & 31);
    long t = i >> 5;
    int hh = (int)(t % nh);
    int s  = (int)(t / nh);
    size_t row = (size_t)s * nh * 64 + (size_t)hh * 64;
    float x1 = X[row + d], x2 = X[row + d + 32];
    float c  = cs[(size_t)s * 64 + d];
    float si = sn[(size_t)s * 64 + d];
    float o1 = x1 * c - x2 * si;     // first half: x1*cos - x2*sin
    float o2 = x2 * c + x1 * si;     // second half: x2*cos + x1*sin
    unsigned short h1 = f2bf(o1);
    hi[row + d] = h1;  lo[row + d] = f2bf(o1 - bf2f(h1));
    unsigned short h2 = f2bf(o2);
    hi[row + d + 32] = h2;  lo[row + d + 32] = f2bf(o2 - bf2f(h2));
}

// V: [S][DK] fp32 -> Vt hi/lo: [DK][S] bf16 (so V^T loads as WMMA A-operand)
__global__ void vtrans_split_kernel(const float* __restrict__ V,
                                    unsigned short* __restrict__ hi,
                                    unsigned short* __restrict__ lo,
                                    int S, int DK) {
    long i = (long)blockIdx.x * blockDim.x + threadIdx.x;
    long n = (long)S * DK;
    if (i >= n) return;
    int s   = (int)(i / DK);
    int cdo = (int)(i - (long)s * DK);
    float f = V[i];
    unsigned short h = f2bf(f);
    size_t o = (size_t)cdo * S + s;
    hi[o] = h;
    lo[o] = f2bf(f - bf2f(h));
}

// ---------------------------------------------------------------------------
// C[M][N] = A[M][K] * B[K][N] + bias, with A,B pre-split to bf16 hi/lo,
// B stored transposed ([N][K]).  3x bf16 WMMA per product for ~fp32 accuracy.
// One wave computes a 16x64 tile. Requires M%16==0, N%64==0, K%32==0.
// ---------------------------------------------------------------------------
__global__ __launch_bounds__(256)
void gemm_bf16x3_kernel(const unsigned short* __restrict__ Ahi,
                        const unsigned short* __restrict__ Alo,
                        const unsigned short* __restrict__ BThi,
                        const unsigned short* __restrict__ BTlo,
                        const float* __restrict__ bias,
                        float* __restrict__ C,
                        int M, int N, int K) {
    const int lane = threadIdx.x & 31;
    const int wid  = (blockIdx.x << 3) + (threadIdx.x >> 5);
    const int nGroups = N >> 6;
    const int mt = wid / nGroups;
    const int nt = wid - mt * nGroups;
    if ((mt << 4) >= M) return;                 // wave-uniform
    const int m0 = mt << 4;
    const int n0 = nt << 6;
    const int col = lane & 15;
    const int g   = lane >> 4;

    v8f acc[4];
#pragma unroll
    for (int j = 0; j < 4; ++j) {
        float b = bias ? bias[n0 + (j << 4) + col] : 0.0f;
#pragma unroll
        for (int v = 0; v < 8; ++v) acc[j][v] = b;
    }

    const unsigned short* aHiRow = Ahi + (size_t)(m0 + col) * K;
    const unsigned short* aLoRow = Alo + (size_t)(m0 + col) * K;

    for (int k0 = 0; k0 < K; k0 += 32) {
        // A-operand layout: lane holds row m=col; K chunks {8g..8g+7, 16+8g..16+8g+7}
        V16 aH, aL;
        aH.q[0] = *reinterpret_cast<const u32x4*>(aHiRow + k0 + 8 * g);
        aH.q[1] = *reinterpret_cast<const u32x4*>(aHiRow + k0 + 16 + 8 * g);
        aL.q[0] = *reinterpret_cast<const u32x4*>(aLoRow + k0 + 8 * g);
        aL.q[1] = *reinterpret_cast<const u32x4*>(aLoRow + k0 + 16 + 8 * g);
        __builtin_prefetch(aHiRow + k0 + 32, 0, 1);
#pragma unroll
        for (int j = 0; j < 4; ++j) {
            // B-operand layout: lane holds col n; K chunks {8g..8g+7, 16+8g..16+8g+7}
            const size_t boff = (size_t)(n0 + (j << 4) + col) * K + k0 + 8 * g;
            V16 bH, bL;
            bH.q[0] = reinterpret_cast<const u32x4*>(BThi + boff)[0];
            bH.q[1] = reinterpret_cast<const u32x4*>(BThi + boff)[2];
            bL.q[0] = reinterpret_cast<const u32x4*>(BTlo + boff)[0];
            bL.q[1] = reinterpret_cast<const u32x4*>(BTlo + boff)[2];
            acc[j] = wmma_bf16(aH, bH, acc[j]);
            acc[j] = wmma_bf16(aH, bL, acc[j]);
            acc[j] = wmma_bf16(aL, bH, acc[j]);
        }
    }

    // C layout: lane holds col n=col, rows v + 8g
#pragma unroll
    for (int j = 0; j < 4; ++j)
#pragma unroll
        for (int v = 0; v < 8; ++v)
            C[(size_t)(m0 + v + 8 * g) * N + n0 + (j << 4) + col] = acc[j][v];
}

// ---------------------------------------------------------------------------
// Sliding-window causal attention with sinks.
// One wave per (head, 16-query tile).  Computes S^T = K*Q^T so softmax stats
// are per-column (lane-local + one xor-16 pair exchange), then O^T = V^T * P.
// ---------------------------------------------------------------------------
__global__ __launch_bounds__(256)
void attn_swin_kernel(const unsigned short* __restrict__ Qhi,
                      const unsigned short* __restrict__ Qlo,  // [S][NH*64]
                      const unsigned short* __restrict__ Khi,
                      const unsigned short* __restrict__ Klo,  // [S][NKV*64]
                      const unsigned short* __restrict__ Vthi,
                      const unsigned short* __restrict__ Vtlo, // [NKV*64][S]
                      const float* __restrict__ sinks,
                      const int* __restrict__ winp,
                      float* __restrict__ Ctx,                 // [S][NH*64]
                      int S, int NHp, int NKVp) {
    const int lane   = threadIdx.x & 31;
    const int wid    = (blockIdx.x << 3) + (threadIdx.x >> 5);
    const int qtiles = S >> 4;
    const int h      = wid / qtiles;
    if (h >= NHp) return;                       // wave-uniform
    const int qt  = wid - h * qtiles;
    const int q0  = qt << 4;
    const int kvh = h / (NHp / NKVp);           // GQA group
    const int col = lane & 15;
    const int g   = lane >> 4;
    const int W   = *winp;
    const float scale = 0.125f;                 // 1/sqrt(64)
    const int DQ = NHp * 64, DK = NKVp * 64;

    // Q^T as the B-operand (col = query q0+col, K-dim rows contiguous)
    V16 qbH[2], qbL[2];
    {
        const size_t qrow = (size_t)(q0 + col) * DQ + (size_t)h * 64;
#pragma unroll
        for (int ck = 0; ck < 2; ++ck) {
            size_t off = qrow + ck * 32 + 8 * g;
            qbH[ck].q[0] = reinterpret_cast<const u32x4*>(Qhi + off)[0];
            qbH[ck].q[1] = reinterpret_cast<const u32x4*>(Qhi + off)[2];
            qbL[ck].q[0] = reinterpret_cast<const u32x4*>(Qlo + off)[0];
            qbL[ck].q[1] = reinterpret_cast<const u32x4*>(Qlo + off)[2];
        }
    }

    v8f o[4];
#pragma unroll
    for (int dt = 0; dt < 4; ++dt)
#pragma unroll
        for (int v = 0; v < 8; ++v) o[dt][v] = 0.f;
    float m_run = -1e30f, s_run = 0.f;

    int lowk = q0 - (W - 1);
    int t_lo = lowk > 0 ? (lowk >> 5) : 0;
    int t_hi = (q0 + 15) >> 5;

    for (int t = t_lo; t <= t_hi; ++t) {
        const int krow = t << 5;                // 32-key tile
        v8f st[2];
#pragma unroll
        for (int sub = 0; sub < 2; ++sub) {
#pragma unroll
            for (int v = 0; v < 8; ++v) st[sub][v] = 0.f;
            const unsigned short* kb = Khi + (size_t)(krow + sub * 16 + col) * DK + (size_t)kvh * 64;
            const unsigned short* kl = Klo + (size_t)(krow + sub * 16 + col) * DK + (size_t)kvh * 64;
#pragma unroll
            for (int ck = 0; ck < 2; ++ck) {
                V16 aH, aL;     // K-matrix rows as A-operand
                aH.q[0] = *reinterpret_cast<const u32x4*>(kb + ck * 32 + 8 * g);
                aH.q[1] = *reinterpret_cast<const u32x4*>(kb + ck * 32 + 16 + 8 * g);
                aL.q[0] = *reinterpret_cast<const u32x4*>(kl + ck * 32 + 8 * g);
                aL.q[1] = *reinterpret_cast<const u32x4*>(kl + ck * 32 + 16 + 8 * g);
                st[sub] = wmma_bf16(aH, qbH[ck], st[sub]);
                st[sub] = wmma_bf16(aH, qbL[ck], st[sub]);
                st[sub] = wmma_bf16(aL, qbH[ck], st[sub]);
            }
        }

        // scale + window mask (lane holds query q = q0+col, keys v+8g per sub)
        const int q = q0 + col;
        float tm = -1e30f;
        float p0[8], p1[8];
#pragma unroll
        for (int v = 0; v < 8; ++v) {
            int k0i = krow + v + 8 * g;
            int k1i = krow + 16 + v + 8 * g;
            bool ok0 = (q >= k0i) && (q - k0i < W);
            bool ok1 = (q >= k1i) && (q - k1i < W);
            float s0v = ok0 ? st[0][v] * scale : -1e30f;
            float s1v = ok1 ? st[1][v] * scale : -1e30f;
            p0[v] = s0v; p1[v] = s1v;
            tm = fmaxf(tm, fmaxf(s0v, s1v));
        }
        tm = fmaxf(tm, __shfl_xor(tm, 16, 32));
        float m_new = fmaxf(m_run, tm);
        float r = __expf(m_run - m_new);
        float local = 0.f;
#pragma unroll
        for (int v = 0; v < 8; ++v) {
            p0[v] = __expf(p0[v] - m_new);
            p1[v] = __expf(p1[v] - m_new);
            local += p0[v] + p1[v];
        }
        s_run = s_run * r + (local + __shfl_xor(local, 16, 32));
        m_run = m_new;
#pragma unroll
        for (int dt = 0; dt < 4; ++dt)
#pragma unroll
            for (int v = 0; v < 8; ++v) o[dt][v] *= r;

        // Repack P^T (C-layout) into B-operand layout: lane-local, C rows 8g+v already match B k order (recv unused).
        float recv[8];
#pragma unroll
        for (int v = 0; v < 8; ++v) {
            float send = (g == 0) ? p1[v] : p0[v];
            recv[v] = __shfl_xor(send, 16, 32);
        }
        V16 pbH, pbL;
#pragma unroll
        for (int i = 0; i < 8; ++i) {
            float a0 = p0[i];   // keys 8g+0..7
            float a1 = p1[i];   // keys 16+8g+0..7
            unsigned short h0 = f2bf(a0);
            pbH.u[i] = h0;     pbL.u[i] = f2bf(a0 - bf2f(h0));
            unsigned short h1 = f2bf(a1);
            pbH.u[8 + i] = h1; pbL.u[8 + i] = f2bf(a1 - bf2f(h1));
        }

        // O^T += V^T * P  (V^T rows are contiguous in key index)
#pragma unroll
        for (int dt = 0; dt < 4; ++dt) {
            const size_t vrow = (size_t)(kvh * 64 + dt * 16 + col) * S + krow;
            V16 vH, vL;
            vH.q[0] = *reinterpret_cast<const u32x4*>(Vthi + vrow + 8 * g);
            vH.q[1] = *reinterpret_cast<const u32x4*>(Vthi + vrow + 16 + 8 * g);
            vL.q[0] = *reinterpret_cast<const u32x4*>(Vtlo + vrow + 8 * g);
            vL.q[1] = *reinterpret_cast<const u32x4*>(Vtlo + vrow + 16 + 8 * g);
            o[dt] = wmma_bf16(vH, pbH, o[dt]);
            o[dt] = wmma_bf16(vL, pbH, o[dt]);
            o[dt] = wmma_bf16(vH, pbL, o[dt]);
        }
    }

    // Fold attention sink into denominator, normalize, store context.
    float sk = sinks[h];
    float mf = fmaxf(m_run, sk);
    float eo = __expf(m_run - mf);
    float denom = s_run * eo + __expf(sk - mf);
    float oc = eo / denom;
#pragma unroll
    for (int dt = 0; dt < 4; ++dt)
#pragma unroll
        for (int v = 0; v < 8; ++v)
            Ctx[(size_t)(q0 + col) * DQ + (size_t)h * 64 + dt * 16 + v + 8 * g] =
                o[dt][v] * oc;
}

// ---------------------------------------------------------------------------
// Host-side orchestration
// ---------------------------------------------------------------------------
extern "C" void kernel_launch(void* const* d_in, const int* in_sizes, int n_in,
                              void* d_out, int out_size, void* d_ws, size_t ws_size,
                              hipStream_t stream) {
    (void)in_sizes; (void)n_in; (void)out_size; (void)ws_size;
    constexpr int S = 2048, H = 2880, NH = 64, NKV = 8;
    constexpr int DQ = NH * 64;   // 4096
    constexpr int DK = NKV * 64;  // 512

    const float* Xf    = (const float*)d_in[0];
    const float* cosp  = (const float*)d_in[1];
    const float* sinp  = (const float*)d_in[2];
    const float* Wq    = (const float*)d_in[3];
    const float* bq    = (const float*)d_in[4];
    const float* Wk    = (const float*)d_in[5];
    const float* bk    = (const float*)d_in[6];
    const float* Wv    = (const float*)d_in[7];
    const float* bv    = (const float*)d_in[8];
    const float* Wo    = (const float*)d_in[9];
    const float* bo    = (const float*)d_in[10];
    const float* sinks = (const float*)d_in[11];
    const int*   winp  = (const int*)d_in[12];

    char* bump = (char*)d_ws;
    auto alloc = [&](size_t bytes) -> void* {
        void* r = (void*)bump;
        bump += (bytes + 255) & ~(size_t)255;
        return r;
    };
    auto ush = [&](size_t n) { return (unsigned short*)alloc(n * 2); };
    auto flt = [&](size_t n) { return (float*)alloc(n * 4); };

    unsigned short* Xhi  = ush((size_t)S * H);
    unsigned short* Xlo  = ush((size_t)S * H);
    unsigned short* WqTh = ush((size_t)DQ * H);
    unsigned short* WqTl = ush((size_t)DQ * H);
    unsigned short* WkTh = ush((size_t)DK * H);
    unsigned short* WkTl = ush((size_t)DK * H);
    unsigned short* WvTh = ush((size_t)DK * H);
    unsigned short* WvTl = ush((size_t)DK * H);
    unsigned short* WoTh = ush((size_t)H * DQ);
    unsigned short* WoTl = ush((size_t)H * DQ);
    float* Qf = flt((size_t)S * DQ);
    float* Kf = flt((size_t)S * DK);
    float* Vf = flt((size_t)S * DK);
    unsigned short* Qrh = ush((size_t)S * DQ);
    unsigned short* Qrl = ush((size_t)S * DQ);
    unsigned short* Krh = ush((size_t)S * DK);
    unsigned short* Krl = ush((size_t)S * DK);
    unsigned short* Vth = ush((size_t)DK * S);
    unsigned short* Vtl = ush((size_t)DK * S);
    float* Ctx = flt((size_t)S * DQ);
    unsigned short* Ch = ush((size_t)S * DQ);
    unsigned short* Cl = ush((size_t)S * DQ);

    auto eb = [](long n) { return (unsigned)((n + 255) / 256); };

    // 1) split inputs to bf16 hi/lo (weights transposed for B-operand)
    split_bf16_kernel<<<eb((long)S * H), 256, 0, stream>>>(Xf, Xhi, Xlo, (long)S * H);
    split_bf16_T_kernel<<<eb((long)H * DQ), 256, 0, stream>>>(Wq, WqTh, WqTl, H, DQ);
    split_bf16_T_kernel<<<eb((long)H * DK), 256, 0, stream>>>(Wk, WkTh, WkTl, H, DK);
    split_bf16_T_kernel<<<eb((long)H * DK), 256, 0, stream>>>(Wv, WvTh, WvTl, H, DK);
    split_bf16_T_kernel<<<eb((long)DQ * H), 256, 0, stream>>>(Wo, WoTh, WoTl, DQ, H);

    // 2) projections: Q = X*Wq + bq, K = X*Wk + bk, V = X*Wv + bv
    {
        unsigned blocks = (unsigned)(((S / 16) * (DQ / 64) + 7) / 8);
        gemm_bf16x3_kernel<<<blocks, 256, 0, stream>>>(Xhi, Xlo, WqTh, WqTl, bq, Qf, S, DQ, H);
    }
    {
        unsigned blocks = (unsigned)(((S / 16) * (DK / 64) + 7) / 8);
        gemm_bf16x3_kernel<<<blocks, 256, 0, stream>>>(Xhi, Xlo, WkTh, WkTl, bk, Kf, S, DK, H);
        gemm_bf16x3_kernel<<<blocks, 256, 0, stream>>>(Xhi, Xlo, WvTh, WvTl, bv, Vf, S, DK, H);
    }

    // 3) RoPE + split for Q,K; transpose + split for V
    rope_split_kernel<<<eb((long)S * NH * 32), 256, 0, stream>>>(Qf, cosp, sinp, Qrh, Qrl, S, NH);
    rope_split_kernel<<<eb((long)S * NKV * 32), 256, 0, stream>>>(Kf, cosp, sinp, Krh, Krl, S, NKV);
    vtrans_split_kernel<<<eb((long)S * DK), 256, 0, stream>>>(Vf, Vth, Vtl, S, DK);

    // 4) windowed attention with sinks
    {
        unsigned blocks = (unsigned)((NH * (S / 16) + 7) / 8);
        attn_swin_kernel<<<blocks, 256, 0, stream>>>(Qrh, Qrl, Krh, Krl, Vth, Vtl,
                                                     sinks, winp, Ctx, S, NH, NKV);
    }

    // 5) output projection: out = Ctx*Wo + bo
    split_bf16_kernel<<<eb((long)S * DQ), 256, 0, stream>>>(Ctx, Ch, Cl, (long)S * DQ);
    {
        unsigned blocks = (unsigned)(((S / 16) * (H / 64) + 7) / 8);
        gemm_bf16x3_kernel<<<blocks, 256, 0, stream>>>(Ch, Cl, WoTh, WoTl, bo,
                                                       (float*)d_out, S, H, DQ);
    }
}
